// TwinRGCN_34548716929229
// MI455X (gfx1250) — matrix operand order repaired and checked by reading, not compiled, not run on hardware
//
#include <hip/hip_runtime.h>
#include <hip/hip_bf16.h>

// ---------------------------------------------------------------------------
// TwinRGCN for MI455X (gfx1250): bf16 WMMA GEMMs (packed-fragment weights) +
// atomic scatter aggregation.
// ---------------------------------------------------------------------------

typedef __attribute__((ext_vector_type(16))) __bf16 v16bf;
typedef __attribute__((ext_vector_type(8)))  float  v8f;

#define NPAPER  50000
#define NAUTHOR 50000
#define NEDGE   400000
#define DFEAT   256
#define NCLASS  128

static __device__ __forceinline__ unsigned short f32_to_bf16_rte(float f) {
  unsigned int u = __float_as_uint(f);
  u += 0x7FFFu + ((u >> 16) & 1u);   // round-to-nearest-even
  return (unsigned short)(u >> 16);
}

// ---------------------------- utility kernels ------------------------------

__global__ void zero_f32_kernel(float* p, long long n) {
  long long i = (long long)blockIdx.x * blockDim.x + threadIdx.x;
  long long stride = (long long)gridDim.x * blockDim.x;
  for (; i < n; i += stride) p[i] = 0.0f;
}

// plain f32 -> bf16 staging
__global__ void cvt_bf16_kernel(const float* __restrict__ a,
                                unsigned short* __restrict__ out, int n) {
  int i = blockIdx.x * blockDim.x + threadIdx.x;
  if (i >= n) return;
  out[i] = f32_to_bf16_rte(a[i]);
}

// Pack weight W[K=256, N] (optionally summed with b,c: twin combined weights)
// into WMMA B-fragment-major bf16 layout:
//   out[((ntile*8 + kc)*32 + lane)*16 + e] = W[kc*32 + (lane>>4)*16 + e][ntile*16 + (lane&15)]
// so each lane's 16-element fragment is 32 contiguous bytes.
__global__ void pack_weight_bf16_kernel(const float* __restrict__ a,
                                        const float* __restrict__ b,
                                        const float* __restrict__ c,
                                        unsigned short* __restrict__ out,
                                        int N, int total) {
  int idx = blockIdx.x * blockDim.x + threadIdx.x;
  if (idx >= total) return;
  int e    = idx & 15;
  int t    = idx >> 4;
  int lane = t & 31; t >>= 5;
  int kc   = t & 7;  t >>= 3;           // K == 256 -> 8 chunks of 32
  int ntile = t;
  int k = (kc << 5) + ((lane >> 4) << 4) + e;
  int n = (ntile << 4) + (lane & 15);
  long long src = (long long)k * N + n;
  float v = a[src];
  if (b) v += b[src];
  if (c) v += c[src];
  out[idx] = f32_to_bf16_rte(v);
}

__global__ void count_deg_kernel(const int* __restrict__ dst, int e, float* deg) {
  int i = blockIdx.x * blockDim.x + threadIdx.x;
  if (i >= e) return;
  atomicAdd(deg + dst[i], 1.0f);
}

// 64 threads per edge, 4 floats each (float4 gather + 4 f32 atomics)
__global__ void scatter_add_kernel(const float* __restrict__ X,
                                   const int* __restrict__ src,
                                   const int* __restrict__ dst,
                                   int e, float* __restrict__ acc) {
  long long idx = (long long)blockIdx.x * blockDim.x + threadIdx.x;
  if (idx >= (long long)e * 64) return;
  int ed = (int)(idx >> 6);
  int c  = (int)(idx & 63) << 2;
  int s = src[ed], d = dst[ed];
  const float4 v = *(const float4*)(X + (long long)s * DFEAT + c);
  float* o = acc + (long long)d * DFEAT + c;
  atomicAdd(o + 0, v.x);
  atomicAdd(o + 1, v.y);
  atomicAdd(o + 2, v.z);
  atomicAdd(o + 3, v.w);
}

// mean normalize (0 where deg==0) and convert to bf16 staging buffer
__global__ void mean_to_bf16_kernel(const float* __restrict__ acc,
                                    const float* __restrict__ deg,
                                    unsigned short* __restrict__ out, long long n) {
  long long i = (long long)blockIdx.x * blockDim.x + threadIdx.x;
  if (i >= n) return;
  float dg = deg[i >> 8];               // D == 256
  float v = dg > 0.0f ? acc[i] / dg : 0.0f;
  out[i] = f32_to_bf16_rte(v);
}

// ------------------------------ WMMA GEMM ----------------------------------
// C[M,N] = A[M,256](bf16) @ W (fragment-packed bf16) [+ Cin] [+ bias] [relu]
// One wave -> one 16x16 C tile; 8 waves/block share one 16x256 A tile in LDS.
// K fully unrolled: 8 chained v_wmma_f32_16x16x32_bf16 per wave.
// grid = (M/16, N/128), block = 256.
__global__ __launch_bounds__(256)
void wmma_gemm_bf16_kernel(const unsigned short* __restrict__ A,
                           const unsigned short* __restrict__ Bp,
                           const float* __restrict__ Cin,
                           const float* __restrict__ bias,
                           float* __restrict__ Cout,
                           int M, int N, int fuse_relu) {
  __shared__ unsigned short As[16 * 264];   // padded row stride vs bank conflicts
  const int tid   = threadIdx.x;
  const int wave  = tid >> 5;
  const int lane  = tid & 31;
  const int m0    = blockIdx.x << 4;
  const int ntile = (blockIdx.y << 3) + wave;
  const int n0    = ntile << 4;

  // cooperative A-tile load: 16 rows x 256 bf16, 512 uint4 chunks
  for (int i = tid; i < 512; i += 256) {
    int row = i >> 5;
    int col = (i & 31) << 3;
    *(uint4*)&As[row * 264 + col] =
        *(const uint4*)(A + (long long)(m0 + row) * 256 + col);
  }
  __syncthreads();

  const int half = lane >> 4;   // K/M half select per ISA layouts
  const int rc   = lane & 15;   // A row / B,C column

  v8f acc;
  #pragma unroll
  for (int r = 0; r < 8; ++r) acc[r] = 0.0f;
  if (Cin) {
    #pragma unroll
    for (int r = 0; r < 8; ++r)
      acc[r] = Cin[(long long)(m0 + (half << 3) + r) * N + n0 + rc];
  }
  if (bias) {
    float bv = bias[n0 + rc];
    #pragma unroll
    for (int r = 0; r < 8; ++r) acc[r] += bv;
  }

  union Frag { uint4 q[2]; v16bf v; };
  // A: each lane's fragment is 16 contiguous bf16 in the LDS row (16B aligned).
  const unsigned short* arow  = &As[rc * 264 + (half << 4)];
  // B: this wave's packed fragments: 32B contiguous per lane per chunk.
  const unsigned short* bfrag = Bp + (long long)ntile * 4096 + (lane << 4);

  #pragma unroll
  for (int kc = 0; kc < 8; ++kc) {
    Frag fa, fb;
    fa.q[0] = *(const uint4*)(arow + (kc << 5));        // K = kc*32 + half*16 + 0..7
    fa.q[1] = *(const uint4*)(arow + (kc << 5) + 8);    // K = kc*32 + half*16 + 8..15
    fb.q[0] = *(const uint4*)(bfrag + (kc << 9));
    fb.q[1] = *(const uint4*)(bfrag + (kc << 9) + 8);
    acc = __builtin_amdgcn_wmma_f32_16x16x32_bf16(
        /*neg_a=*/false, fa.v, /*neg_b=*/false, fb.v,
        /*c_mod=*/(short)0, acc, /*reuse_a=*/false, /*reuse_b=*/false);
  }

  #pragma unroll
  for (int r = 0; r < 8; ++r) {
    float v = acc[r];
    if (fuse_relu) v = fmaxf(v, 0.0f);
    Cout[(long long)(m0 + (half << 3) + r) * N + n0 + rc] = v;
  }
}

// ---------------------- attention summarize (wave/node) --------------------
__global__ __launch_bounds__(256)
void attn_kernel(const float* __restrict__ H0, const float* __restrict__ H1,
                 const float* __restrict__ H0t, const float* __restrict__ H1t,
                 float* __restrict__ alpha, unsigned short* __restrict__ hb,
                 int nnodes) {
  int node = blockIdx.x * 8 + (threadIdx.x >> 5);
  int lane = threadIdx.x & 31;
  if (node >= nnodes) return;
  long long base = (long long)node * DFEAT + lane * 8;
  float h0[8], h1[8];
  float s0 = 0.f, s1 = 0.f;
  #pragma unroll
  for (int j = 0; j < 8; ++j) {
    h0[j] = H0[base + j];
    h1[j] = H1[base + j];
    s0 += h0[j] * H0t[base + j];
    s1 += h1[j] * H1t[base + j];
  }
  #pragma unroll
  for (int off = 16; off > 0; off >>= 1) {
    s0 += __shfl_xor(s0, off, 32);
    s1 += __shfl_xor(s1, off, 32);
  }
  // TEMP == 1.0
  float mx = fmaxf(s0, s1);
  float e0 = __expf(s0 - mx), e1 = __expf(s1 - mx);
  float inv = 1.0f / (e0 + e1);
  float a0 = e0 * inv, a1 = e1 * inv;
  if (lane == 0) {
    alpha[node]          = a0;   // alpha[0, i]
    alpha[nnodes + node] = a1;   // alpha[1, i]
  }
  #pragma unroll
  for (int j = 0; j < 8; ++j)
    hb[base + j] = f32_to_bf16_rte(a0 * h0[j] + a1 * h1[j]);
}

// ------------------------------ host driver --------------------------------

extern "C" void kernel_launch(void* const* d_in, const int* in_sizes, int n_in,
                              void* d_out, int out_size, void* d_ws, size_t ws_size,
                              hipStream_t stream) {
  (void)in_sizes; (void)n_in; (void)out_size; (void)ws_size;
  const float* x_paper     = (const float*)d_in[0];
  const float* emb_author  = (const float*)d_in[1];
  const float* l0_rootP_W  = (const float*)d_in[2];
  const float* l0_rootP_b  = (const float*)d_in[3];
  const float* l0_rootA_W  = (const float*)d_in[4];
  const float* l0_rootA_b  = (const float*)d_in[5];
  const float* l0_writes_W = (const float*)d_in[6];
  const float* l0_rev_W    = (const float*)d_in[7];
  const float* l0_cites_W  = (const float*)d_in[8];
  const float* l1_rootP_W  = (const float*)d_in[9];
  const float* l1_rootP_b  = (const float*)d_in[10];
  const float* l1_writes_W = (const float*)d_in[13];
  const float* l1_cites_W  = (const float*)d_in[15];
  const float* out_W       = (const float*)d_in[16];
  const float* out_b       = (const float*)d_in[17];
  const int* writes_src = (const int*)d_in[18];
  const int* writes_dst = (const int*)d_in[19];
  const int* rev_src    = (const int*)d_in[20];
  const int* rev_dst    = (const int*)d_in[21];
  const int* cites_src  = (const int*)d_in[22];
  const int* cites_dst  = (const int*)d_in[23];

  // ------------------------- workspace layout ------------------------------
  char* ws = (char*)d_ws;
  const long long FB  = (long long)NPAPER * DFEAT * sizeof(float);
  const long long HBy = (long long)NPAPER * DFEAT * sizeof(unsigned short);
  size_t off = 0;
  float* AGG = (float*)(ws + off); off += FB;   // one reusable f32 agg buffer
  float* P0  = (float*)(ws + off); off += FB;   // h0 paper (main)
  float* P0t = (float*)(ws + off); off += FB;   // h0 paper (twin)
  float* A0  = (float*)(ws + off); off += FB;   // h0 author (main)
  float* P1  = (float*)(ws + off); off += FB;   // h1 paper (main)
  float* P1t = (float*)(ws + off); off += FB;   // h1 paper (twin)
  unsigned short* SA = (unsigned short*)(ws + off); off += HBy;  // bf16 A staging
  float* DEGW = (float*)(ws + off); off += (long long)NPAPER  * 4;
  float* DEGR = (float*)(ws + off); off += (long long)NAUTHOR * 4;
  float* DEGC = (float*)(ws + off); off += (long long)NPAPER  * 4;
  const int WSLOT = 256 * 256;                  // packed bf16 weight slot
  unsigned short* WB = (unsigned short*)(ws + off); off += (long long)11 * WSLOT * 2;
  unsigned short* rootP0b = WB + 0 * WSLOT;
  unsigned short* rootA0b = WB + 1 * WSLOT;
  unsigned short* wrt0b   = WB + 2 * WSLOT;
  unsigned short* rev0b   = WB + 3 * WSLOT;
  unsigned short* cit0b   = WB + 4 * WSLOT;
  unsigned short* twnP0b  = WB + 5 * WSLOT;
  unsigned short* rootP1b = WB + 6 * WSLOT;
  unsigned short* wrt1b   = WB + 7 * WSLOT;
  unsigned short* cit1b   = WB + 8 * WSLOT;
  unsigned short* twnP1b  = WB + 9 * WSLOT;
  unsigned short* outWb   = WB + 10 * WSLOT;

  // --------------------------- helper launchers ----------------------------
  auto pack = [&](const float* a, const float* b, const float* c,
                  unsigned short* o, int N) {
    int total = 256 * N;   // K * N fragment elements
    pack_weight_bf16_kernel<<<dim3((total + 255) / 256), dim3(256), 0, stream>>>(
        a, b, c, o, N, total);
  };
  auto stage = [&](const float* x) {    // f32 [50000,256] -> bf16 SA
    cvt_bf16_kernel<<<dim3((NPAPER * DFEAT + 255) / 256), dim3(256), 0, stream>>>(
        x, SA, NPAPER * DFEAT);
  };
  auto gemm = [&](const unsigned short* A, const unsigned short* Bw,
                  const float* Cin, const float* bias, float* Cout,
                  int N, int relu) {
    dim3 grid(NPAPER / 16, N / 128);
    wmma_gemm_bf16_kernel<<<grid, dim3(256), 0, stream>>>(A, Bw, Cin, bias, Cout,
                                                          NPAPER, N, relu);
  };
  auto agg = [&](const float* X, const int* src, const int* dst, const float* deg) {
    long long n = (long long)NPAPER * DFEAT;
    zero_f32_kernel<<<dim3(2048), dim3(256), 0, stream>>>(AGG, n);
    long long t = (long long)NEDGE * 64;
    scatter_add_kernel<<<dim3((unsigned)((t + 255) / 256)), dim3(256), 0, stream>>>(
        X, src, dst, NEDGE, AGG);
    mean_to_bf16_kernel<<<dim3((unsigned)((n + 255) / 256)), dim3(256), 0, stream>>>(
        AGG, deg, SA, n);
  };

  // ------------------- weight prep (bf16, fragment-packed) -----------------
  pack(l0_rootP_W,  nullptr, nullptr, rootP0b, 256);
  pack(l0_rootA_W,  nullptr, nullptr, rootA0b, 256);
  pack(l0_writes_W, nullptr, nullptr, wrt0b,   256);
  pack(l0_rev_W,    nullptr, nullptr, rev0b,   256);
  pack(l0_cites_W,  nullptr, nullptr, cit0b,   256);
  pack(l0_rootP_W, l0_writes_W, l0_cites_W, twnP0b, 256);   // twin combined W
  pack(l1_rootP_W,  nullptr, nullptr, rootP1b, 256);
  pack(l1_writes_W, nullptr, nullptr, wrt1b,   256);
  pack(l1_cites_W,  nullptr, nullptr, cit1b,   256);
  pack(l1_rootP_W, l1_writes_W, l1_cites_W, twnP1b, 256);
  pack(out_W, nullptr, nullptr, outWb, NCLASS);

  // ------------------------- in-degrees (static) ---------------------------
  zero_f32_kernel<<<dim3(256), dim3(256), 0, stream>>>(DEGW, 3LL * NPAPER);
  count_deg_kernel<<<dim3((NEDGE + 255) / 256), dim3(256), 0, stream>>>(writes_dst, NEDGE, DEGW);
  count_deg_kernel<<<dim3((NEDGE + 255) / 256), dim3(256), 0, stream>>>(rev_dst,    NEDGE, DEGR);
  count_deg_kernel<<<dim3((NEDGE + 255) / 256), dim3(256), 0, stream>>>(cites_dst,  NEDGE, DEGC);

  // ------------------------------ layer 0 ----------------------------------
  stage(x_paper);
  gemm(SA, rootP0b, nullptr, l0_rootP_b, P0, 256, 0);
  agg(emb_author, writes_src, writes_dst, DEGW);
  gemm(SA, wrt0b, P0, nullptr, P0, 256, 0);
  agg(x_paper, cites_src, cites_dst, DEGC);
  gemm(SA, cit0b, P0, nullptr, P0, 256, 1);          // P0 = relu(...)

  stage(emb_author);
  gemm(SA, rootA0b, nullptr, l0_rootA_b, A0, 256, 0);
  agg(x_paper, rev_src, rev_dst, DEGR);
  gemm(SA, rev0b, A0, nullptr, A0, 256, 1);          // A0 = relu(...)

  stage(x_paper);                                    // twin starts from same feats
  gemm(SA, twnP0b, nullptr, l0_rootP_b, P0t, 256, 1);

  // ------------------------------ layer 1 ----------------------------------
  // (author outputs at layer 1 feed nothing downstream -> skipped)
  stage(P0);
  gemm(SA, rootP1b, nullptr, l1_rootP_b, P1, 256, 0);
  agg(A0, writes_src, writes_dst, DEGW);
  gemm(SA, wrt1b, P1, nullptr, P1, 256, 0);
  agg(P0, cites_src, cites_dst, DEGC);
  gemm(SA, cit1b, P1, nullptr, P1, 256, 1);          // P1 = relu(...)

  stage(P0t);
  gemm(SA, twnP1b, nullptr, l1_rootP_b, P1t, 256, 1);

  // ------------------- summarize attention + output GEMM -------------------
  float* logits = (float*)d_out;                       // [50000,128]
  float* alpha  = logits + (long long)NPAPER * NCLASS; // [2,50000]
  attn_kernel<<<dim3(NPAPER / 8), dim3(256), 0, stream>>>(P0, P1, P0t, P1t,
                                                          alpha, SA, NPAPER);
  gemm(SA, outWb, nullptr, out_b, logits, NCLASS, 0);
}
